// CapibaraJAXSSM_5317169512703
// MI455X (gfx1250) — matrix operand order repaired and checked by reading, not compiled, not run on hardware
//
#include <hip/hip_runtime.h>
#include <cstdint>

typedef __bf16 bf16;
typedef __attribute__((ext_vector_type(16))) __bf16 v16bf;
typedef __attribute__((ext_vector_type(8)))  __bf16 bf16x8;
typedef __attribute__((ext_vector_type(8)))  float  v8f;

#define HID    2048
#define BATCHN 32
#define SEQN   512
#define MROWS  (BATCHN*SEQN)   // 16384
#define LN_EPS 1e-6f

// LDS tile pitch: 32 bf16 of data padded to 40 elements (80 B) per row so that
// the 16 rows read by one ds_load_b128 fragment land in distinct bank groups.
#define TPITCH 40

// ---------------------------------------------------------------------------
// CDNA5 primitives
// ---------------------------------------------------------------------------
__device__ __forceinline__ v8f wmma_bf16(v16bf a, v16bf b, v8f c) {
  // 8 args: (neg_a, A, neg_b, B, c_mod, C, reuse_a, reuse_b)
  return __builtin_amdgcn_wmma_f32_16x16x32_bf16(false, a, false, b, (short)0, c,
                                                 false, false);
}

// Async global->LDS copy, 16 bytes per lane, tracked by ASYNCcnt.
__device__ __forceinline__ void async_copy_b128(uint32_t lds_addr, const void* gaddr) {
  asm volatile("global_load_async_to_lds_b128 %0, %1, off"
               :: "v"(lds_addr), "v"((uint64_t)(uintptr_t)gaddr) : "memory");
}
__device__ __forceinline__ void wait_asynccnt_3() {
  asm volatile("s_wait_asynccnt 0x3" ::: "memory");
}
__device__ __forceinline__ void wait_asynccnt_0() {
  asm volatile("s_wait_asynccnt 0x0" ::: "memory");
}

// A-matrix fragment, 16x32 bf16 (ISA 7.12.2): lanes 0-15 row m=lane hold
// K=[0..7],[16..23]; lanes 16-31 hold K=[8..15],[24..31].
// rowp points at row start (global K-row or LDS tile row); laneHi = lane>>4.
__device__ __forceinline__ v16bf load_a_frag(const bf16* rowp, int laneHi) {
  const bf16* p = rowp + laneHi * 8;
  bf16x8 lo = *(const bf16x8*)p;
  bf16x8 hi = *(const bf16x8*)(p + 16);
  return __builtin_shufflevector(lo, hi, 0,1,2,3,4,5,6,7,8,9,10,11,12,13,14,15);
}

// B-matrix fragment, 32x16 bf16: lane column n=lane&15; elements 0-7 <-> K=laneHi*8+e,
// elements 8-15 <-> K=16+laneHi*8+(e-8); p points at rowp + laneHi*8.
__device__ __forceinline__ v16bf load_b_frag(const bf16* p) {
  bf16x8 lo = *(const bf16x8*)p;
  bf16x8 hi = *(const bf16x8*)(p + 16);
  return __builtin_shufflevector(lo, hi, 0,1,2,3,4,5,6,7,8,9,10,11,12,13,14,15);
}

// ---------------------------------------------------------------------------
// Conversion kernels
// ---------------------------------------------------------------------------
__global__ __launch_bounds__(256) void cvt_f32_bf16(const float* __restrict__ src,
                                                    bf16* __restrict__ dst, int n) {
  int i = blockIdx.x * 256 + threadIdx.x;
  if (i < n) dst[i] = (bf16)src[i];
}

// Transposing convert: src f32 [K,N] -> dst bf16 [N,K] (for W_out, which the
// reference applies un-transposed; our GEMM is NT so we pre-transpose it).
__global__ __launch_bounds__(256) void cvt_f32_bf16_T(const float* __restrict__ src,
                                                      bf16* __restrict__ dst, int n) {
  int i = blockIdx.x * 256 + threadIdx.x;
  if (i < n) {
    int k = i >> 11;
    int c = i & (HID - 1);
    dst[(size_t)c * HID + k] = (bf16)src[i];
  }
}

__global__ __launch_bounds__(256) void copy_final_state(const float* __restrict__ s,
                                                        float* __restrict__ dst, int n) {
  int i = blockIdx.x * 256 + threadIdx.x;
  if (i < n) dst[i] = s[i];
}

// ---------------------------------------------------------------------------
// LayerNorm + bf16 convert + [B,S,H] -> [t*B+b] row reorder.
// ---------------------------------------------------------------------------
__global__ __launch_bounds__(256) void ln_to_bf16(const float* __restrict__ x,
                                                  const float* __restrict__ gamma,
                                                  const float* __restrict__ beta,
                                                  bf16* __restrict__ out) {
  const int r = blockIdx.x;            // r = t*32 + b
  const int t = r >> 5, b = r & 31;
  const float* src = x + ((size_t)b * SEQN + t) * HID;
  bf16* dst = out + (size_t)r * HID;
  const int tid = threadIdx.x;

  float4 v0 = *(const float4*)(src + tid * 8);
  float4 v1 = *(const float4*)(src + tid * 8 + 4);
  float s  = v0.x + v0.y + v0.z + v0.w + v1.x + v1.y + v1.z + v1.w;
  float ss = v0.x*v0.x + v0.y*v0.y + v0.z*v0.z + v0.w*v0.w +
             v1.x*v1.x + v1.y*v1.y + v1.z*v1.z + v1.w*v1.w;
  #pragma unroll
  for (int off = 16; off > 0; off >>= 1) {
    s  += __shfl_down(s,  off, 32);
    ss += __shfl_down(ss, off, 32);
  }
  __shared__ float sm[8], sm2[8];
  if ((tid & 31) == 0) { sm[tid >> 5] = s; sm2[tid >> 5] = ss; }
  __syncthreads();
  float S = 0.f, SS = 0.f;
  #pragma unroll
  for (int i = 0; i < 8; i++) { S += sm[i]; SS += sm2[i]; }
  const float mu   = S * (1.f / HID);
  const float var  = SS * (1.f / HID) - mu * mu;
  const float rstd = rsqrtf(var + LN_EPS);

  const int h = tid * 8;
  float4 g0 = *(const float4*)(gamma + h);
  float4 g1 = *(const float4*)(gamma + h + 4);
  float4 be0 = *(const float4*)(beta + h);
  float4 be1 = *(const float4*)(beta + h + 4);
  bf16x8 o;
  o[0] = (bf16)((v0.x - mu) * rstd * g0.x + be0.x);
  o[1] = (bf16)((v0.y - mu) * rstd * g0.y + be0.y);
  o[2] = (bf16)((v0.z - mu) * rstd * g0.z + be0.z);
  o[3] = (bf16)((v0.w - mu) * rstd * g0.w + be0.w);
  o[4] = (bf16)((v1.x - mu) * rstd * g1.x + be1.x);
  o[5] = (bf16)((v1.y - mu) * rstd * g1.y + be1.y);
  o[6] = (bf16)((v1.z - mu) * rstd * g1.z + be1.z);
  o[7] = (bf16)((v1.w - mu) * rstd * g1.w + be1.w);
  *(bf16x8*)(dst + h) = o;
}

// ---------------------------------------------------------------------------
// Batched NT GEMM with async-LDS double buffering:
//   Out[M,N] = X[M,K] * W[N,K]^T   (bf16 in, f32 accumulate)
// Block = 8 waves = 64(M) x 128(N); wave = 2x2 WMMA tiles (32x32).
// K-step 32: stage X tile (64x32) + W tile (128x32) into LDS with
// global_load_async_to_lds_b128, double-buffered; fragments read via ds_load_b128.
// OUT==0: f32 store.  OUT==1: bf16 store.  OUT==2: f32 + bias + [t*B+b]->[b,s] remap.
// ---------------------------------------------------------------------------
template <int OUT>
__global__ __launch_bounds__(256) void gemm_bf16_nt(const bf16* __restrict__ X,
                                                    const bf16* __restrict__ Wt,
                                                    float* __restrict__ outF,
                                                    bf16* __restrict__ outB,
                                                    const float* __restrict__ bias,
                                                    int M, int N, int K) {
  __shared__ __align__(16) bf16 sX[2][64 * TPITCH];
  __shared__ __align__(16) bf16 sW[2][128 * TPITCH];

  const int tid = threadIdx.x;
  const int lane = tid & 31, wave = tid >> 5;
  const int laneHi = lane >> 4, lane15 = lane & 15;
  const int waveM = wave >> 2, waveN = wave & 3;
  const int mBlk = blockIdx.y * 64;
  const int nBlk = blockIdx.x * 128;

  // per-thread async-copy slots: 16B per lane per instruction
  const int cRow = tid >> 2, cSeg = tid & 3;         // X: 64 rows x 4 segs
  const bf16* xg  = X  + (size_t)(mBlk + cRow) * K + cSeg * 8;
  const bf16* wgA = Wt + (size_t)(nBlk + cRow) * K + cSeg * 8;        // W rows 0..63
  const bf16* wgB = Wt + (size_t)(nBlk + 64 + cRow) * K + cSeg * 8;   // W rows 64..127
  const uint32_t xl  = (uint32_t)(uintptr_t)&sX[0][0] +
                       (uint32_t)(cRow * TPITCH * 2 + cSeg * 16);
  const uint32_t wlA = (uint32_t)(uintptr_t)&sW[0][0] +
                       (uint32_t)(cRow * TPITCH * 2 + cSeg * 16);
  const uint32_t wlB = wlA + (uint32_t)(64 * TPITCH * 2);
  const uint32_t xStride = 64 * TPITCH * 2;    // bytes per X buffer
  const uint32_t wStride = 128 * TPITCH * 2;   // bytes per W buffer

  auto issue = [&](int p, int kk) {
    async_copy_b128(xl  + (uint32_t)p * xStride, xg  + kk);
    async_copy_b128(wlA + (uint32_t)p * wStride, wgA + kk);
    async_copy_b128(wlB + (uint32_t)p * wStride, wgB + kk);
  };

  v8f acc00 = {}, acc01 = {}, acc10 = {}, acc11 = {};

  const int nk = K / 32;
  issue(0, 0);
  for (int ik = 0; ik < nk; ik++) {
    const int p = ik & 1;
    if (ik + 1 < nk) {
      issue(p ^ 1, (ik + 1) * 32);
      wait_asynccnt_3();           // previous batch (3 async ops) complete
    } else {
      wait_asynccnt_0();
    }
    __syncthreads();

    const bf16* xb = &sX[p][0];
    const bf16* wb = &sW[p][0];
    v16bf a0 = load_a_frag(xb + (waveM * 32 + lane15) * TPITCH, laneHi);
    v16bf a1 = load_a_frag(xb + (waveM * 32 + 16 + lane15) * TPITCH, laneHi);
    v16bf b0 = load_b_frag(wb + (waveN * 32 + lane15) * TPITCH + laneHi * 8);
    v16bf b1 = load_b_frag(wb + (waveN * 32 + 16 + lane15) * TPITCH + laneHi * 8);
    acc00 = wmma_bf16(a0, b0, acc00);
    acc01 = wmma_bf16(a0, b1, acc01);
    acc10 = wmma_bf16(a1, b0, acc10);
    acc11 = wmma_bf16(a1, b1, acc11);
    __syncthreads();               // buffer p free for reuse two iterations out
  }

  const int m0 = mBlk + waveM * 32;
  const int n0 = nBlk + waveN * 32;
  #pragma unroll
  for (int i = 0; i < 2; i++) {
    #pragma unroll
    for (int j = 0; j < 2; j++) {
      v8f acc = (i == 0) ? (j == 0 ? acc00 : acc01) : (j == 0 ? acc10 : acc11);
      const int mb = m0 + i * 16, nb = n0 + j * 16;
      #pragma unroll
      for (int r = 0; r < 8; r++) {
        const int row = mb + r + laneHi * 8;
        const int col = nb + lane15;
        if (OUT == 0) {
          outF[(size_t)row * N + col] = acc[r];
        } else if (OUT == 1) {
          outB[(size_t)row * N + col] = (bf16)acc[r];
        } else {
          const int t = row >> 5, b = row & 31;          // row = t*32 + b
          outF[((size_t)b * SEQN + t) * N + col] = acc[r] + bias[col];
        }
      }
    }
  }
}

// ---------------------------------------------------------------------------
// One recurrence step: s_out[32,H] = s_in[32,H] @ A^T + u_t ; also emit bf16
// copy into the state history. Wave: one 16x16 tile; block = 32x64; grid = 32.
// ---------------------------------------------------------------------------
__global__ __launch_bounds__(256) void scan_step(const float* __restrict__ s_in,
                                                 const bf16* __restrict__ Abf,
                                                 const float* __restrict__ u_t,
                                                 float* __restrict__ s_out,
                                                 bf16* __restrict__ hist) {
  const int tid = threadIdx.x;
  const int lane = tid & 31, wave = tid >> 5;
  const int laneHi = lane >> 4, lane15 = lane & 15;
  const int waveM = wave >> 2, waveN = wave & 3;
  const int m0 = waveM * 16;
  const int n0 = blockIdx.x * 64 + waveN * 16;

  v8f acc = {};
  const float* sr = s_in + (size_t)(m0 + lane15) * HID;
  const bf16* wr = Abf + (size_t)(n0 + lane15) * HID + laneHi * 8;

  for (int kk = 0; kk < HID; kk += 32) {
    __builtin_prefetch(wr + kk + 512, 0, 0);   // global_prefetch_b8
    const float* p = sr + kk + laneHi * 8;
    float4 f0 = *(const float4*)(p);
    float4 f1 = *(const float4*)(p + 4);
    float4 f2 = *(const float4*)(p + 16);
    float4 f3 = *(const float4*)(p + 20);
    v16bf a;
    a[0]  = (bf16)f0.x; a[1]  = (bf16)f0.y; a[2]  = (bf16)f0.z; a[3]  = (bf16)f0.w;
    a[4]  = (bf16)f1.x; a[5]  = (bf16)f1.y; a[6]  = (bf16)f1.z; a[7]  = (bf16)f1.w;
    a[8]  = (bf16)f2.x; a[9]  = (bf16)f2.y; a[10] = (bf16)f2.z; a[11] = (bf16)f2.w;
    a[12] = (bf16)f3.x; a[13] = (bf16)f3.y; a[14] = (bf16)f3.z; a[15] = (bf16)f3.w;
    v16bf b = load_b_frag(wr + kk);
    acc = wmma_bf16(a, b, acc);
  }

  #pragma unroll
  for (int r = 0; r < 8; r++) {
    const int row = m0 + r + laneHi * 8;
    const int col = n0 + lane15;
    const size_t idx = (size_t)row * HID + col;
    const float v = acc[r] + u_t[idx];
    s_out[idx] = v;
    hist[idx] = (bf16)v;
  }
}

// ---------------------------------------------------------------------------
// Host-side orchestration
// ---------------------------------------------------------------------------
extern "C" void kernel_launch(void* const* d_in, const int* in_sizes, int n_in,
                              void* d_out, int out_size, void* d_ws, size_t ws_size,
                              hipStream_t stream) {
  const float* x     = (const float*)d_in[0];   // [32,512,2048]
  const float* state = (const float*)d_in[1];   // [32,2048]
  const float* A     = (const float*)d_in[2];   // [2048,2048]
  const float* Bm    = (const float*)d_in[3];
  const float* Cm    = (const float*)d_in[4];
  const float* Wo    = (const float*)d_in[5];
  const float* bout  = (const float*)d_in[6];   // [2048]
  const float* gamma = (const float*)d_in[7];
  const float* beta  = (const float*)d_in[8];
  float* out = (float*)d_out;                   // [32,512,2048] ++ [32,2048]

  char* ws = (char*)d_ws;
  size_t off = 0;
  auto carve = [&](size_t bytes) {
    void* p = ws + off;
    off += (bytes + 255) & ~(size_t)255;
    return p;
  };
  const size_t WN = (size_t)HID * HID;          // 4M weight elements
  bf16* Abf  = (bf16*)carve(WN * 2);
  bf16* Bbf  = (bf16*)carve(WN * 2);
  bf16* Cbf  = (bf16*)carve(WN * 2);
  bf16* Wtbf = (bf16*)carve(WN * 2);            // transposed W_out
  bf16* xn   = (bf16*)carve((size_t)MROWS * HID * 2);   // LN(x); reused as y
  float* u   = (float*)carve((size_t)MROWS * HID * 4);  // xn @ B^T
  float* sb0 = (float*)carve((size_t)BATCHN * HID * 4);
  float* sb1 = (float*)carve((size_t)BATCHN * HID * 4);
  bf16* Sall = (bf16*)carve((size_t)MROWS * HID * 2);   // state history (bf16)
  (void)ws_size; (void)in_sizes; (void)n_in; (void)out_size;

  // 1) weights -> bf16 (W_out also transposed)
  const int wgrid = (int)(WN / 256);
  cvt_f32_bf16<<<wgrid, 256, 0, stream>>>(A,  Abf, (int)WN);
  cvt_f32_bf16<<<wgrid, 256, 0, stream>>>(Bm, Bbf, (int)WN);
  cvt_f32_bf16<<<wgrid, 256, 0, stream>>>(Cm, Cbf, (int)WN);
  cvt_f32_bf16_T<<<wgrid, 256, 0, stream>>>(Wo, Wtbf, (int)WN);

  // 2) LayerNorm -> bf16, rows reordered to [t*32+b]
  ln_to_bf16<<<MROWS, 256, 0, stream>>>(x, gamma, beta, xn);

  // 3) u = xn @ B^T  (f32 out; feeds the f32 recurrence)
  dim3 gBig(HID / 128, MROWS / 64);             // (16, 256)
  gemm_bf16_nt<0><<<gBig, 256, 0, stream>>>(xn, Bbf, u, nullptr, nullptr,
                                            MROWS, HID, HID);

  // 4) sequential scan: s_t = s_{t-1} @ A^T + u_t  (512 dependent launches)
  const float* sprev = state;
  for (int t = 0; t < SEQN; t++) {
    float* snext = (t & 1) ? sb1 : sb0;
    scan_step<<<HID / 64, 256, 0, stream>>>(sprev, Abf,
                                            u + (size_t)t * BATCHN * HID, snext,
                                            Sall + (size_t)t * BATCHN * HID);
    sprev = snext;
  }

  // 5) y = S_all @ C^T  (bf16 out, reusing xn buffer)
  gemm_bf16_nt<1><<<gBig, 256, 0, stream>>>(Sall, Cbf, nullptr, xn, nullptr,
                                            MROWS, HID, HID);

  // 6) out = y @ W_out + b_out, with [t*32+b] -> [b,s] remap into d_out
  gemm_bf16_nt<2><<<gBig, 256, 0, stream>>>(xn, Wtbf, out, nullptr, bout,
                                            MROWS, HID, HID);

  // 7) final state
  copy_final_state<<<(BATCHN * HID) / 256, 256, 0, stream>>>(
      sprev, out + (size_t)BATCHN * SEQN * HID, BATCHN * HID);
}
